// DecoderRNN_65292092833987
// MI455X (gfx1250) — compile-verified, matrix operand order checked by reading, not run on hardware
//
#include <hip/hip_runtime.h>
#include <math.h>

// ---------------- problem constants ----------------
#define BB 40
#define SS 512
#define TT 20
#define VV 78864
#define EE 300
#define HH 512
#define AA 300
#define ELM 812          // E + H
#define GG 2048          // 4*H
#define NK 1624          // ELM + A + H
#define KP 320           // K=300 padded to 320 for bf16 WMMA (10 chunks of 32)
#define NT 4929          // VV / 16

typedef __attribute__((ext_vector_type(16))) __bf16 v16bf;
typedef __attribute__((ext_vector_type(8)))  float  v8f;

__device__ __forceinline__ float sigm(float x) { return 1.0f / (1.0f + __expf(-x)); }

// ---------------- setup kernels ----------------

// out (C x R) = transpose of in (R x C)
__global__ void k_transpose(const float* __restrict__ in, float* __restrict__ out,
                            int R, int C) {
    size_t total = (size_t)R * C;
    size_t stride = (size_t)gridDim.x * blockDim.x;
    for (size_t idx = (size_t)blockIdx.x * blockDim.x + threadIdx.x; idx < total; idx += stride) {
        size_t r = idx / C, c = idx % C;
        out[c * (size_t)R + r] = in[idx];
    }
}

__global__ void k_fill_f32(float* p, float v, size_t n) {
    size_t stride = (size_t)gridDim.x * blockDim.x;
    for (size_t i = (size_t)blockIdx.x * blockDim.x + threadIdx.x; i < n; i += stride) p[i] = v;
}

__global__ void k_fill_u16(unsigned short* p, unsigned short v, size_t n) {
    size_t stride = (size_t)gridDim.x * blockDim.x;
    for (size_t i = (size_t)blockIdx.x * blockDim.x + threadIdx.x; i < n; i += stride) p[i] = v;
}

// Wout (V x 300) f32  ->  WoutBF (V x 320) bf16, zero padded
__global__ void k_wout_cvt(const float* __restrict__ Wout, __bf16* __restrict__ Wbf) {
    size_t total = (size_t)VV * KP;
    size_t stride = (size_t)gridDim.x * blockDim.x;
    for (size_t idx = (size_t)blockIdx.x * blockDim.x + threadIdx.x; idx < total; idx += stride) {
        size_t v = idx / KP, k = idx % KP;
        float x = (k < AA) ? Wout[v * AA + k] : 0.0f;
        Wbf[idx] = (__bf16)x;
    }
}

// K[b,s,h] = sum_k enc_out[b,s,k] * Wenc[h,k]    (one block per (b,s))
__global__ __launch_bounds__(256) void k_K(const float* __restrict__ enc_out,
                                           const float* __restrict__ WencT,
                                           float* __restrict__ Kb) {
    __shared__ float s_r[HH];
    int bs = blockIdx.x, j = threadIdx.x;
    const float* row = enc_out + (size_t)bs * HH;
    s_r[j] = row[j]; s_r[j + 256] = row[j + 256];
    __syncthreads();
    for (int h = j; h < HH; h += 256) {
        float acc = 0.0f;
        for (int k = 0; k < HH; ++k) acc += s_r[k] * WencT[(size_t)k * HH + h];
        Kb[(size_t)bs * HH + h] = acc;
    }
}

// ---------------- per-step kernels ----------------

struct RnnPtrs {
    const int*   inputs;
    const float* embed;
    const float* Wih1T; const float* Whh1T; const float* b1;
    const float* Wih2T; const float* Whh2T; const float* b2;
    const float* WihT;  const float* WhhT;  const float* brnn;
    const float* WinT;  const float* bin_l; const float* pos_embed;
    const float* WdecT;
    float *h1, *c1, *h2, *c2, *h, *c;
    float *datt, *sread;           // previous-step values (read only here)
    float *e_buf, *Q;
};

// one block per batch row: embed lookup + 3 chained LSTMs + dec_in GEMM + Q
__global__ __launch_bounds__(512) void k_rnn(RnnPtrs P, int t) {
    __shared__ float s_e[304], s_h1[HH], s_c1[HH], s_h2[HH], s_c2[HH],
                     s_h[HH], s_c[HH], s_da[304], s_sr[HH], s_x[HH];
    const int b = blockIdx.x, j = threadIdx.x;

    int sym = (t == 0) ? 2 : P.inputs[b * TT + (t - 1)];
    for (int k = j; k < EE; k += 512) s_e[k] = P.embed[(size_t)sym * EE + k];
    s_h1[j] = P.h1[b * HH + j]; s_c1[j] = P.c1[b * HH + j];
    s_h2[j] = P.h2[b * HH + j]; s_c2[j] = P.c2[b * HH + j];
    s_h[j]  = P.h [b * HH + j]; s_c[j]  = P.c [b * HH + j];
    for (int k = j; k < AA; k += 512) s_da[k] = P.datt[b * AA + k];
    s_sr[j] = P.sread[b * HH + j];
    __syncthreads();

    // ---- LSTM1 : x = e (300), h = h1 ----
    float g0 = P.b1[j], g1 = P.b1[HH + j], g2 = P.b1[2 * HH + j], g3 = P.b1[3 * HH + j];
    for (int k = 0; k < EE; ++k) {
        float x = s_e[k]; const float* w = P.Wih1T + (size_t)k * GG + j;
        g0 += x * w[0]; g1 += x * w[HH]; g2 += x * w[2 * HH]; g3 += x * w[3 * HH];
    }
    for (int k = 0; k < HH; ++k) {
        float x = s_h1[k]; const float* w = P.Whh1T + (size_t)k * GG + j;
        g0 += x * w[0]; g1 += x * w[HH]; g2 += x * w[2 * HH]; g3 += x * w[3 * HH];
    }
    float c1n = sigm(g1) * s_c1[j] + sigm(g0) * tanhf(g2);
    float h1n = sigm(g3) * tanhf(c1n);
    __syncthreads();
    s_c1[j] = c1n; s_h1[j] = h1n;
    __syncthreads();

    // ---- LSTM2 : x = h1, h = h2 ----
    g0 = P.b2[j]; g1 = P.b2[HH + j]; g2 = P.b2[2 * HH + j]; g3 = P.b2[3 * HH + j];
    for (int k = 0; k < HH; ++k) {
        float x = s_h1[k]; const float* w = P.Wih2T + (size_t)k * GG + j;
        g0 += x * w[0]; g1 += x * w[HH]; g2 += x * w[2 * HH]; g3 += x * w[3 * HH];
    }
    for (int k = 0; k < HH; ++k) {
        float x = s_h2[k]; const float* w = P.Whh2T + (size_t)k * GG + j;
        g0 += x * w[0]; g1 += x * w[HH]; g2 += x * w[2 * HH]; g3 += x * w[3 * HH];
    }
    float c2n = sigm(g1) * s_c2[j] + sigm(g0) * tanhf(g2);
    float h2n = sigm(g3) * tanhf(c2n);
    __syncthreads();
    s_c2[j] = c2n; s_h2[j] = h2n;
    __syncthreads();

    // ---- dec_in = [e, h2, datt, sread] @ Win.T + bin + pos_embed[t] ----
    float acc = P.bin_l[j] + P.pos_embed[t * HH + j];
    for (int k = 0; k < EE; ++k) acc += s_e[k]  * P.WinT[(size_t)k * HH + j];
    for (int k = 0; k < HH; ++k) acc += s_h2[k] * P.WinT[(size_t)(EE + k) * HH + j];
    for (int k = 0; k < AA; ++k) acc += s_da[k] * P.WinT[(size_t)(ELM + k) * HH + j];
    for (int k = 0; k < HH; ++k) acc += s_sr[k] * P.WinT[(size_t)(ELM + AA + k) * HH + j];
    s_x[j] = acc;
    __syncthreads();

    // ---- main LSTM : x = dec_in, h ----
    g0 = P.brnn[j]; g1 = P.brnn[HH + j]; g2 = P.brnn[2 * HH + j]; g3 = P.brnn[3 * HH + j];
    for (int k = 0; k < HH; ++k) {
        float x = s_x[k]; const float* w = P.WihT + (size_t)k * GG + j;
        g0 += x * w[0]; g1 += x * w[HH]; g2 += x * w[2 * HH]; g3 += x * w[3 * HH];
    }
    for (int k = 0; k < HH; ++k) {
        float x = s_h[k]; const float* w = P.WhhT + (size_t)k * GG + j;
        g0 += x * w[0]; g1 += x * w[HH]; g2 += x * w[2 * HH]; g3 += x * w[3 * HH];
    }
    float cn = sigm(g1) * s_c[j] + sigm(g0) * tanhf(g2);
    float hn = sigm(g3) * tanhf(cn);
    __syncthreads();
    s_c[j] = cn; s_h[j] = hn;
    __syncthreads();

    // ---- Q = h @ Wdec.T ----
    float q = 0.0f;
    for (int k = 0; k < HH; ++k) q += s_h[k] * P.WdecT[(size_t)k * HH + j];
    P.Q[b * HH + j] = q;

    // ---- write back state ----
    P.h1[b * HH + j] = s_h1[j]; P.c1[b * HH + j] = s_c1[j];
    P.h2[b * HH + j] = s_h2[j]; P.c2[b * HH + j] = s_c2[j];
    P.h [b * HH + j] = s_h[j];  P.c [b * HH + j] = s_c[j];
    for (int k = j; k < EE; k += 512) P.e_buf[b * EE + k] = s_e[k];
}

// logits[b,s] = masked ? -inf : sum_h tanh(Q[b,h]+K[b,s,h]+his[b,s]*whis[h]+bias[h])*vt[h]
__global__ __launch_bounds__(256) void k_attn(const float* __restrict__ Q,
                                              const float* __restrict__ Kb,
                                              const float* __restrict__ his,
                                              const float* __restrict__ whis,
                                              const float* __restrict__ vt,
                                              const float* __restrict__ bias,
                                              const unsigned char* __restrict__ enc_mask,
                                              float* __restrict__ logits) {
    int idx = blockIdx.x * blockDim.x + threadIdx.x;
    if (idx >= BB * SS) return;
    if (enc_mask[idx]) { logits[idx] = -INFINITY; return; }
    int b = idx >> 9;
    float hb = his[idx];
    const float* Kr = Kb + (size_t)idx * HH;
    const float* Qr = Q + b * HH;
    float acc = 0.0f;
    for (int h = 0; h < HH; ++h)
        acc += tanhf(Qr[h] + Kr[h] + hb * whis[h] + bias[h]) * vt[h];
    logits[idx] = acc;
}

// softmax + cov partial + his update + context + sread   (one block per batch row)
__global__ __launch_bounds__(256) void k_softmax(const float* __restrict__ logits,
                                                 const float* __restrict__ enc_out,
                                                 const int* __restrict__ enc_in,
                                                 const int* __restrict__ inputs,
                                                 float* __restrict__ attn,
                                                 float* __restrict__ his,
                                                 float* __restrict__ ctx,
                                                 float* __restrict__ sread,
                                                 float* __restrict__ cov_part, int t) {
    __shared__ float red[256];
    __shared__ float s_a[SS];
    int b = blockIdx.x, j = threadIdx.x;

    float m = -INFINITY;
    for (int s = j; s < SS; s += 256) m = fmaxf(m, logits[b * SS + s]);
    red[j] = m; __syncthreads();
    for (int st = 128; st > 0; st >>= 1) { if (j < st) red[j] = fmaxf(red[j], red[j + st]); __syncthreads(); }
    float mx = red[0]; __syncthreads();

    float sum = 0.0f;
    for (int s = j; s < SS; s += 256) { float ex = __expf(logits[b * SS + s] - mx); s_a[s] = ex; sum += ex; }
    red[j] = sum; __syncthreads();
    for (int st = 128; st > 0; st >>= 1) { if (j < st) red[j] += red[j + st]; __syncthreads(); }
    float inv = 1.0f / red[0]; __syncthreads();

    float covp = 0.0f;
    for (int s = j; s < SS; s += 256) {
        float a = s_a[s] * inv; s_a[s] = a; attn[b * SS + s] = a;
        float ho = his[b * SS + s];
        covp += fminf(a, ho);
        his[b * SS + s] = 0.5f * (a + ho);
    }
    red[j] = covp; __syncthreads();
    for (int st = 128; st > 0; st >>= 1) { if (j < st) red[j] += red[j + st]; __syncthreads(); }
    if (j == 0) cov_part[b] = red[0];
    __syncthreads();

    int tok = inputs[b * TT + t];
    for (int d = j; d < HH; d += 256) {
        float ca = 0.0f, ra = 0.0f;
        const float* eo = enc_out + (size_t)b * SS * HH + d;
        for (int s = 0; s < SS; ++s) {
            float v = eo[(size_t)s * HH];
            float a = s_a[s];
            ca += a * v;
            if (enc_in[b * SS + s] == tok) ra += a * v;
        }
        ctx[b * HH + d] = ca;
        sread[b * HH + d] = ra;
    }
}

// datt = [ctx, h] @ Watt.T + batt ; pc; score_e ; bf16 pack of datt  (one block per b)
__global__ __launch_bounds__(320) void k_datt(const float* __restrict__ ctx,
                                              const float* __restrict__ st_h,
                                              const float* __restrict__ WattT,
                                              const float* __restrict__ batt,
                                              const float* __restrict__ Wcopy_attn,
                                              const float* __restrict__ Wcopy_in,
                                              const float* __restrict__ scale,
                                              const float* __restrict__ e_buf,
                                              const float* __restrict__ st_h2,
                                              const float* __restrict__ attn,
                                              float* __restrict__ st_datt,
                                              __bf16* __restrict__ datt_bf,
                                              float* __restrict__ score_e) {
    __shared__ float s_x[2 * HH];
    __shared__ float s_d[304];
    __shared__ float red[512];
    int b = blockIdx.x, j = threadIdx.x;
    for (int k = j; k < HH; k += 320) { s_x[k] = ctx[b * HH + k]; s_x[HH + k] = st_h[b * HH + k]; }
    __syncthreads();
    if (j < AA) {
        float acc = batt[j];
        for (int k = 0; k < 2 * HH; ++k) acc += s_x[k] * WattT[(size_t)k * AA + j];
        s_d[j] = acc;
        st_datt[b * AA + j] = acc;
        datt_bf[(size_t)b * KP + j] = (__bf16)acc;
    }
    __syncthreads();
    float p = 0.0f;
    for (int k = j; k < AA; k += 320) p += s_d[k] * Wcopy_attn[k];
    for (int k = j; k < AA; k += 320) p += e_buf[b * AA + k] * Wcopy_in[k];
    for (int k = j; k < HH; k += 320) p += st_h2[b * HH + k] * Wcopy_in[AA + k];
    red[j] = p;
    if (j < 512 - 320) red[320 + j] = 0.0f;
    __syncthreads();
    for (int st = 256; st > 0; st >>= 1) { if (j < st) red[j] += red[j + st]; __syncthreads(); }
    float pc = red[0] + scale[0];
    float pc2 = pc * pc;
    for (int s = j; s < SS; s += 320) score_e[b * SS + s] = attn[b * SS + s] * pc2;
}

__global__ void k_cov(float* cov, const float* part) {
    if (blockIdx.x == 0 && threadIdx.x == 0) {
        float s = *cov;
        for (int i = 0; i < BB; ++i) s += part[i];
        *cov = s;
    }
}

// ---------------- the WMMA output GEMM ----------------
// out[b, t, v] = datt[b,:] . Wout[v,:] + bout[v]
// one wave per 16-wide column of V; 3 M-tiles (48 padded rows); K = 320 (10 chunks)

__device__ __forceinline__ v16bf load_ab(const __bf16* __restrict__ row, int hi) {
    union { v16bf v; uint4 q[2]; } u;
    u.q[0] = *(const uint4*)(row + hi * 8);        // K = hi*8 .. hi*8+7
    u.q[1] = *(const uint4*)(row + 16 + hi * 8);   // K = 16+hi*8 .. 16+hi*8+7
    return u.v;
}

__global__ __launch_bounds__(256) void k_wout_wmma(const __bf16* __restrict__ Wbf,
                                                   const __bf16* __restrict__ datt_bf,
                                                   const float* __restrict__ bout,
                                                   float* __restrict__ out, int t) {
    int wave = threadIdx.x >> 5;
    int lane = threadIdx.x & 31;
    int ntile = blockIdx.x * 8 + wave;          // wave-uniform
    if (ntile >= NT) return;                    // whole wave exits together -> EXEC stays all-1s
    int n0 = ntile * 16;
    int r  = lane & 15;
    int hi = lane >> 4;

    const __bf16* brow = Wbf + (size_t)(n0 + r) * KP;
    const __bf16* a0r  = datt_bf + (size_t)(r)      * KP;
    const __bf16* a1r  = datt_bf + (size_t)(16 + r) * KP;
    const __bf16* a2r  = datt_bf + (size_t)(32 + r) * KP;

    v8f acc0 = {}, acc1 = {}, acc2 = {};
    #pragma unroll
    for (int kc = 0; kc < KP / 32; ++kc) {
        int kb = kc * 32;
        v16bf bm = load_ab(brow + kb, hi);
        v16bf a0 = load_ab(a0r + kb, hi);
        v16bf a1 = load_ab(a1r + kb, hi);
        v16bf a2 = load_ab(a2r + kb, hi);
        acc0 = __builtin_amdgcn_wmma_f32_16x16x32_bf16(false, a0, false, bm, (short)0, acc0, false, false);
        acc1 = __builtin_amdgcn_wmma_f32_16x16x32_bf16(false, a1, false, bm, (short)0, acc1, false, false);
        acc2 = __builtin_amdgcn_wmma_f32_16x16x32_bf16(false, a2, false, bm, (short)0, acc2, false, false);
    }
    float bb = bout[n0 + r];
    size_t col = (size_t)n0 + r;
    #pragma unroll
    for (int i = 0; i < 8; ++i) {
        int m0 = i + hi * 8;                               // rows 0..15
        out[((size_t)m0 * TT + t) * VV + col] = acc0[i] + bb;
        int m1 = 16 + m0;                                  // rows 16..31
        out[((size_t)m1 * TT + t) * VV + col] = acc1[i] + bb;
        int m2 = 32 + m0;                                  // rows 32..47 (only <40 valid)
        if (m2 < BB) out[((size_t)m2 * TT + t) * VV + col] = acc2[i] + bb;
    }
}

// deterministic scatter-add: one thread per batch row, sequential over S
__global__ void k_scatter(float* __restrict__ out, const int* __restrict__ enc_in,
                          const float* __restrict__ score_e, int t) {
    int b = blockIdx.x * blockDim.x + threadIdx.x;
    if (b >= BB) return;
    float* row = out + ((size_t)b * TT + t) * VV;
    for (int s = 0; s < SS; ++s) row[enc_in[b * SS + s]] += score_e[b * SS + s];
}

__global__ __launch_bounds__(256) void k_argmax(const float* __restrict__ out,
                                                float* __restrict__ preds, int t) {
    __shared__ float rv[256];
    __shared__ int   ri[256];
    int b = blockIdx.x, j = threadIdx.x;
    const float* row = out + ((size_t)b * TT + t) * VV;
    float best = -INFINITY; int bi = VV;
    for (int v = j; v < VV; v += 256) { float x = row[v]; if (x > best) { best = x; bi = v; } }
    rv[j] = best; ri[j] = bi; __syncthreads();
    for (int st = 128; st > 0; st >>= 1) {
        if (j < st) {
            float xv = rv[j + st]; int xi = ri[j + st];
            if (xv > rv[j] || (xv == rv[j] && xi < ri[j])) { rv[j] = xv; ri[j] = xi; }
        }
        __syncthreads();
    }
    if (j == 0) preds[b * TT + t] = (float)ri[0];
}

__global__ void k_finish(float* out, const float* cov) {
    if (blockIdx.x == 0 && threadIdx.x == 0)
        out[(size_t)BB * TT * VV + BB * TT] = *cov;
}

// ---------------- host launcher ----------------
extern "C" void kernel_launch(void* const* d_in, const int* in_sizes, int n_in,
                              void* d_out, int out_size, void* d_ws, size_t ws_size,
                              hipStream_t stream) {
    (void)in_sizes; (void)n_in; (void)out_size; (void)ws_size;

    const int*           enc_in    = (const int*)  d_in[0];
    const float*         enc_out   = (const float*)d_in[1];
    const unsigned char* enc_mask  = (const unsigned char*)d_in[2];   // numpy bool (1 byte)
    const int*           inputs    = (const int*)  d_in[3];
    const float*         embed     = (const float*)d_in[4];
    const float*         Wih1      = (const float*)d_in[5];
    const float*         Whh1      = (const float*)d_in[6];
    const float*         b1        = (const float*)d_in[7];
    const float*         Wih2      = (const float*)d_in[8];
    const float*         Whh2      = (const float*)d_in[9];
    const float*         b2        = (const float*)d_in[10];
    const float*         Wih       = (const float*)d_in[11];
    const float*         Whh       = (const float*)d_in[12];
    const float*         brnn      = (const float*)d_in[13];
    const float*         Wdec      = (const float*)d_in[14];
    const float*         Wenc      = (const float*)d_in[15];
    const float*         Whis      = (const float*)d_in[16];  // (512,1) -> flat 512
    const float*         VT        = (const float*)d_in[17];  // (1,512) -> flat 512
    const float*         attn_bias = (const float*)d_in[18];
    const float*         Win       = (const float*)d_in[19];
    const float*         bin_l     = (const float*)d_in[20];
    const float*         Wout      = (const float*)d_in[21];
    const float*         bout      = (const float*)d_in[22];
    const float*         Watt      = (const float*)d_in[23];
    const float*         batt      = (const float*)d_in[24];
    const float*         Wcopy_attn= (const float*)d_in[25];
    const float*         Wcopy_in  = (const float*)d_in[26];
    const float*         scale     = (const float*)d_in[27];
    const float*         pos_embed = (const float*)d_in[28];

    float* out = (float*)d_out;

    // workspace carve-up (all L2-resident working set: ~121 MB)
    char* base = (char*)d_ws; size_t off = 0;
    auto alloc = [&](size_t bytes) -> void* {
        off = (off + 255) & ~(size_t)255;
        void* p = base + off; off += bytes; return p;
    };
    float*  Wih1T  = (float*)alloc(sizeof(float) * (size_t)EE * GG);
    float*  Whh1T  = (float*)alloc(sizeof(float) * (size_t)HH * GG);
    float*  Wih2T  = (float*)alloc(sizeof(float) * (size_t)HH * GG);
    float*  Whh2T  = (float*)alloc(sizeof(float) * (size_t)HH * GG);
    float*  WihT   = (float*)alloc(sizeof(float) * (size_t)HH * GG);
    float*  WhhT   = (float*)alloc(sizeof(float) * (size_t)HH * GG);
    float*  WinT   = (float*)alloc(sizeof(float) * (size_t)NK * HH);
    float*  WdecT  = (float*)alloc(sizeof(float) * (size_t)HH * HH);
    float*  WattT  = (float*)alloc(sizeof(float) * (size_t)2 * HH * AA);
    float*  WencT  = (float*)alloc(sizeof(float) * (size_t)HH * HH);
    __bf16* WoutBF = (__bf16*)alloc(2 * (size_t)VV * KP);
    __bf16* dattBF = (__bf16*)alloc(2 * (size_t)48 * KP);
    float*  Kb     = (float*)alloc(sizeof(float) * (size_t)BB * SS * HH);
    float*  st_h1  = (float*)alloc(sizeof(float) * BB * HH);
    float*  st_c1  = (float*)alloc(sizeof(float) * BB * HH);
    float*  st_h2  = (float*)alloc(sizeof(float) * BB * HH);
    float*  st_c2  = (float*)alloc(sizeof(float) * BB * HH);
    float*  st_h   = (float*)alloc(sizeof(float) * BB * HH);
    float*  st_c   = (float*)alloc(sizeof(float) * BB * HH);
    float*  st_da  = (float*)alloc(sizeof(float) * BB * AA);
    float*  st_sr  = (float*)alloc(sizeof(float) * BB * HH);
    float*  his    = (float*)alloc(sizeof(float) * BB * SS);
    float*  attn   = (float*)alloc(sizeof(float) * BB * SS);
    float*  score_e= (float*)alloc(sizeof(float) * BB * SS);
    float*  logits = (float*)alloc(sizeof(float) * BB * SS);
    float*  Q      = (float*)alloc(sizeof(float) * BB * HH);
    float*  ctx    = (float*)alloc(sizeof(float) * BB * HH);
    float*  e_buf  = (float*)alloc(sizeof(float) * BB * EE);
    float*  covp   = (float*)alloc(sizeof(float) * BB);
    float*  cov    = (float*)alloc(sizeof(float));

    // -------- setup (once per launch) --------
    k_transpose<<<1024, 256, 0, stream>>>(Wih1, Wih1T, GG, EE);
    k_transpose<<<1024, 256, 0, stream>>>(Whh1, Whh1T, GG, HH);
    k_transpose<<<1024, 256, 0, stream>>>(Wih2, Wih2T, GG, HH);
    k_transpose<<<1024, 256, 0, stream>>>(Whh2, Whh2T, GG, HH);
    k_transpose<<<1024, 256, 0, stream>>>(Wih,  WihT,  GG, HH);
    k_transpose<<<1024, 256, 0, stream>>>(Whh,  WhhT,  GG, HH);
    k_transpose<<<1024, 256, 0, stream>>>(Win,  WinT,  HH, NK);
    k_transpose<<<1024, 256, 0, stream>>>(Wdec, WdecT, HH, HH);
    k_transpose<<<1024, 256, 0, stream>>>(Watt, WattT, AA, 2 * HH);
    k_transpose<<<1024, 256, 0, stream>>>(Wenc, WencT, HH, HH);
    k_wout_cvt<<<4096, 256, 0, stream>>>(Wout, WoutBF);
    k_K<<<BB * SS, 256, 0, stream>>>(enc_out, WencT, Kb);

    k_fill_f32<<<64, 256, 0, stream>>>(st_h1, 0.0f, (size_t)BB * HH);
    k_fill_f32<<<64, 256, 0, stream>>>(st_c1, 0.0f, (size_t)BB * HH);
    k_fill_f32<<<64, 256, 0, stream>>>(st_h2, 0.0f, (size_t)BB * HH);
    k_fill_f32<<<64, 256, 0, stream>>>(st_c2, 0.0f, (size_t)BB * HH);
    k_fill_f32<<<64, 256, 0, stream>>>(st_h,  0.0f, (size_t)BB * HH);
    k_fill_f32<<<64, 256, 0, stream>>>(st_c,  0.0f, (size_t)BB * HH);
    k_fill_f32<<<64, 256, 0, stream>>>(st_da, 0.0f, (size_t)BB * AA);
    k_fill_f32<<<64, 256, 0, stream>>>(st_sr, 0.0f, (size_t)BB * HH);
    k_fill_f32<<<64, 256, 0, stream>>>(his, 1.0f / SS, (size_t)BB * SS);
    k_fill_f32<<<1, 32, 0, stream>>>(cov, 0.0f, 1);
    k_fill_u16<<<64, 256, 0, stream>>>((unsigned short*)dattBF, 0, (size_t)48 * KP);

    RnnPtrs P;
    P.inputs = inputs; P.embed = embed;
    P.Wih1T = Wih1T; P.Whh1T = Whh1T; P.b1 = b1;
    P.Wih2T = Wih2T; P.Whh2T = Whh2T; P.b2 = b2;
    P.WihT = WihT;   P.WhhT = WhhT;   P.brnn = brnn;
    P.WinT = WinT;   P.bin_l = bin_l; P.pos_embed = pos_embed;
    P.WdecT = WdecT;
    P.h1 = st_h1; P.c1 = st_c1; P.h2 = st_h2; P.c2 = st_c2; P.h = st_h; P.c = st_c;
    P.datt = st_da; P.sread = st_sr;
    P.e_buf = e_buf; P.Q = Q;

    float* preds = out + (size_t)BB * TT * VV;

    // -------- sequential decode steps --------
    for (int t = 0; t < TT; ++t) {
        k_rnn<<<BB, 512, 0, stream>>>(P, t);
        k_attn<<<(BB * SS) / 256, 256, 0, stream>>>(Q, Kb, his, Whis, VT, attn_bias,
                                                    enc_mask, logits);
        k_softmax<<<BB, 256, 0, stream>>>(logits, enc_out, enc_in, inputs,
                                          attn, his, ctx, st_sr, covp, t);
        k_datt<<<BB, 320, 0, stream>>>(ctx, st_h, WattT, batt, Wcopy_attn, Wcopy_in,
                                       scale, e_buf, st_h2, attn, st_da, dattBF, score_e);
        k_cov<<<1, 32, 0, stream>>>(cov, covp);
        k_wout_wmma<<<(NT + 7) / 8, 256, 0, stream>>>(WoutBF, dattBF, bout, out, t);
        k_scatter<<<1, 64, 0, stream>>>(out, enc_in, score_e, t);
        k_argmax<<<BB, 256, 0, stream>>>(out, preds, t);
    }
    k_finish<<<1, 1, 0, stream>>>(out, cov);
}
